// LSTMP_87986700026255
// MI455X (gfx1250) — matrix operand order repaired and checked by reading, not compiled, not run on hardware
//
#include <hip/hip_runtime.h>
#include <hip/hip_bf16.h>
#include <stdint.h>

// ---------------- problem dims ----------------
#define TT   1024
#define NB   64
#define IN   512
#define HID  1024
#define PR   512

#define SCAN_BLOCKS  32
#define SCAN_THREADS 256   // 8 waves/block -> 256 waves total

// LDS staging strides (halfwords), padded +8 so row base banks differ (260/516 ≡ 4 mod 64)
#define P1_STRIDE 520      // 512 + 8
#define P2_STRIDE 1032     // 1024 + 8

// ---------------- workspace byte offsets ----------------
#define OFF_XB   ((size_t)0)
#define OFF_WIH  ((size_t)67108864)
#define OFF_WHH  ((size_t)71303168)
#define OFF_WHR  ((size_t)75497472)
#define OFF_XG   ((size_t)76546048)
#define OFF_C    ((size_t)613416960)
#define OFF_HB   ((size_t)613679104)
#define OFF_HF   ((size_t)613744640)
#define OFF_BAR  ((size_t)613875712)

// ---------------- vector types ----------------
typedef __attribute__((ext_vector_type(16))) __bf16         v16bf;
typedef __attribute__((ext_vector_type(8)))  float          v8f;
typedef __attribute__((ext_vector_type(8)))  unsigned short v8us;
typedef __attribute__((ext_vector_type(16))) unsigned short v16us;
typedef __attribute__((ext_vector_type(4)))  int            v4i;

// ---------------- async global->LDS availability ----------------
#if defined(__has_builtin)
#  if __has_builtin(__builtin_amdgcn_global_load_async_to_lds_b128)
#    define HAVE_ASYNC_LDS 1
#  else
#    define HAVE_ASYNC_LDS 0
#  endif
#else
#  define HAVE_ASYNC_LDS 0
#endif

// ---------------- helpers ----------------
__device__ __forceinline__ unsigned short f32_to_bf16(float f) {
    unsigned u = __builtin_bit_cast(unsigned, f);
    unsigned r = u + 0x7FFFu + ((u >> 16) & 1u);   // round-to-nearest-even
    return (unsigned short)(r >> 16);
}
__device__ __forceinline__ float bf16_to_f32(unsigned short h) {
    return __builtin_bit_cast(float, (unsigned)h << 16);
}
__device__ __forceinline__ float fast_sigmoid(float x) {
    return 1.0f / (1.0f + __expf(-x));
}
__device__ __forceinline__ float fast_tanh(float x) {
    float ax = fabsf(x);
    float e  = __expf(-2.0f * ax);
    float t  = (1.0f - e) / (1.0f + e);
    return copysignf(t, x);
}
// Build a 16-element bf16 fragment from two contiguous 16B runs.
__device__ __forceinline__ v16bf frag16(const unsigned short* p0,
                                        const unsigned short* p1) {
    v8us lo = *(const v8us*)p0;
    v8us hi = *(const v8us*)p1;
    v16us r = __builtin_shufflevector(lo, hi,
        0,1,2,3,4,5,6,7,8,9,10,11,12,13,14,15);
    return __builtin_bit_cast(v16bf, r);
}
__device__ __forceinline__ v8f wmma_bf16(v16bf a, v16bf b, v8f c) {
    return __builtin_amdgcn_wmma_f32_16x16x32_bf16(
        false, a, false, b, (short)0, c, false, false);
}

// Stage one 16B chunk global -> LDS (async path if available).
__device__ __forceinline__ void stage_chunk(const unsigned short* g,
                                            unsigned short* l) {
#if HAVE_ASYNC_LDS
    __builtin_amdgcn_global_load_async_to_lds_b128(
        (__attribute__((address_space(1))) v4i*)g,
        (__attribute__((address_space(3))) v4i*)l, 0, 0);
#else
    *(uint4*)l = *(const uint4*)g;
#endif
}
__device__ __forceinline__ void stage_wait() {
#if HAVE_ASYNC_LDS
#  if __has_builtin(__builtin_amdgcn_s_wait_asynccnt)
    __builtin_amdgcn_s_wait_asynccnt(0);
#  else
    asm volatile("s_wait_asynccnt 0" ::: "memory");
#  endif
#endif
}

// agent-scope grid barrier (persistent-kernel sync)
__device__ __forceinline__ void grid_barrier(unsigned* cnt, unsigned* gen,
                                             unsigned nblk) {
    __syncthreads();
    if (threadIdx.x == 0) {
        __threadfence();
        unsigned g = __hip_atomic_load(gen, __ATOMIC_RELAXED,
                                       __HIP_MEMORY_SCOPE_AGENT);
        unsigned a = __hip_atomic_fetch_add(cnt, 1u, __ATOMIC_ACQ_REL,
                                            __HIP_MEMORY_SCOPE_AGENT);
        if (a == nblk - 1u) {
            __hip_atomic_store(cnt, 0u, __ATOMIC_RELAXED,
                               __HIP_MEMORY_SCOPE_AGENT);
            __hip_atomic_store(gen, g + 1u, __ATOMIC_RELEASE,
                               __HIP_MEMORY_SCOPE_AGENT);
        } else {
            while (__hip_atomic_load(gen, __ATOMIC_ACQUIRE,
                                     __HIP_MEMORY_SCOPE_AGENT) == g) {
                __builtin_amdgcn_s_sleep(2);
            }
        }
        __threadfence();
    }
    __syncthreads();
}

// ---------------- conversion / init kernels ----------------
__global__ void k_f32_to_bf16(const float* __restrict__ in,
                              unsigned short* __restrict__ out, int n) {
    int i = blockIdx.x * blockDim.x + threadIdx.x;
    if (i < n) out[i] = f32_to_bf16(__builtin_nontemporal_load(in + i));
}
__global__ void k_init_state(const float* __restrict__ h0,
                             const float* __restrict__ c0,
                             unsigned short* __restrict__ hb,
                             float* __restrict__ c,
                             unsigned* __restrict__ bar) {
    int i = blockIdx.x * blockDim.x + threadIdx.x;
    if (i < NB * PR)  hb[i] = f32_to_bf16(h0[i]);
    if (i < NB * HID) c[i]  = c0[i];
    if (i < 2)        bar[i] = 0u;
}

// ---------------- kernel 1: x_gates = x @ w_ih^T + b ----------------
__global__ void __launch_bounds__(256)
k_gates_gemm(const unsigned short* __restrict__ xb,
             const unsigned short* __restrict__ wb,
             const float* __restrict__ bias,
             unsigned short* __restrict__ xg) {
    const int lane  = threadIdx.x & 31;
    const int wave  = (blockIdx.x * blockDim.x + threadIdx.x) >> 5;
    const int nwav  = (gridDim.x * blockDim.x) >> 5;
    const int m16   = lane & 15;
    const int half  = lane >> 4;
    const int abase = half * 8;    // A: lanes16-31 start at K=8 (ISA 7.12.2)
    const int bbase = half * 8;    // B: same K order as A (lanes16-31 start at K=8)
    const int ntask = (TT * NB / 16) * (HID / 16);   // 4096 * 64

    for (int task = wave; task < ntask; task += nwav) {
        const int mt = task >> 6;
        const int jt = task & 63;
        v8f acc[4] = {v8f{0}, v8f{0}, v8f{0}, v8f{0}};
        const unsigned short* arow = xb + (size_t)(mt * 16 + m16) * IN;
        #pragma unroll 4
        for (int k0 = 0; k0 < IN; k0 += 32) {
            v16bf a = frag16(arow + k0 + abase, arow + k0 + 16 + abase);
            #pragma unroll
            for (int g = 0; g < 4; ++g) {
                const unsigned short* brow =
                    wb + (size_t)(g * HID + jt * 16 + m16) * IN + k0 + bbase;
                acc[g] = wmma_bf16(a, frag16(brow, brow + 16), acc[g]);
            }
        }
        #pragma unroll
        for (int g = 0; g < 4; ++g) {
            const float bv = bias[g * HID + jt * 16 + m16];
            v8us out;
            #pragma unroll
            for (int r = 0; r < 8; ++r) out[r] = f32_to_bf16(acc[g][r] + bv);
            __builtin_nontemporal_store(
                out, (v8us*)(xg + ((size_t)task * 4 + g) * 256 + lane * 8));
        }
    }
}

// ---------------- kernel 2: persistent recurrent scan ----------------
__global__ void __launch_bounds__(SCAN_THREADS)
k_scan(const unsigned short* __restrict__ xg,
       const unsigned short* __restrict__ whh,
       const unsigned short* __restrict__ whr,
       unsigned short* __restrict__ hb,     // [64,512] bf16
       float* __restrict__ c,               // [64,1024]
       unsigned short* __restrict__ hfull,  // [64,1024] bf16
       float* __restrict__ y,               // [T,64,512]
       float* __restrict__ hn,              // [64,512]
       float* __restrict__ cn,              // [64,1024]
       unsigned* __restrict__ bar) {
    __shared__ unsigned short smem[16 * P2_STRIDE];  // 33 KB, reused per phase

    const int lane  = threadIdx.x & 31;
    const int wave  = (blockIdx.x * blockDim.x + threadIdx.x) >> 5; // 0..255
    const int m16   = lane & 15;
    const int half  = lane >> 4;
    const int abase = half * 8;
    const int bbase = half * 8;

    for (int t = 0; t < TT; ++t) {
        // ---- stage this block's shared h slab (16 rows x 512) into LDS ----
        {
            const int mtb = blockIdx.x >> 3;          // uniform: == each wave's mt
            const unsigned short* gsrc = hb + (size_t)mtb * 16 * PR;
            #pragma unroll
            for (int i = 0; i < 4; ++i) {
                int ci  = threadIdx.x + i * SCAN_THREADS;   // 0..1023 chunks
                int row = ci >> 6, col = (ci & 63) * 8;
                stage_chunk(gsrc + row * PR + col, smem + row * P1_STRIDE + col);
            }
            stage_wait();
        }
        __syncthreads();

        // ---- phase 1: gates = xg + h @ w_hh^T ; cell update ----
        {
            const int mt = wave >> 6;   // 0..3  batch tile
            const int jt = wave & 63;   // 0..63 hidden tile
            const size_t tb = (((size_t)t * 4 + mt) * 64 + jt) * 4;
            if (t + 1 < TT) {  // prefetch next step's gate fragments
                __builtin_prefetch(xg + (tb + (size_t)4 * 64 * 4) * 256 + lane * 8, 0, 1);
            }
            v8f acc[4];
            #pragma unroll
            for (int g = 0; g < 4; ++g) {
                v8us f = __builtin_nontemporal_load(
                    (const v8us*)(xg + (tb + g) * 256 + lane * 8));
                #pragma unroll
                for (int r = 0; r < 8; ++r) acc[g][r] = bf16_to_f32(f[r]);
            }
            const unsigned short* arow = smem + m16 * P1_STRIDE;  // LDS A slab
            #pragma unroll 4
            for (int k0 = 0; k0 < PR; k0 += 32) {
                v16bf a = frag16(arow + k0 + abase, arow + k0 + 16 + abase);
                #pragma unroll
                for (int g = 0; g < 4; ++g) {
                    const unsigned short* brow =
                        whh + (size_t)(g * HID + jt * 16 + m16) * PR + k0 + bbase;
                    acc[g] = wmma_bf16(a, frag16(brow, brow + 16), acc[g]);
                }
            }
            const int col = jt * 16 + m16;
            #pragma unroll
            for (int r = 0; r < 8; ++r) {
                const int n = mt * 16 + half * 8 + r;
                const size_t ci = (size_t)n * HID + col;
                float iv = fast_sigmoid(acc[0][r]);
                float fv = fast_sigmoid(acc[1][r]);
                float gv = fast_tanh(acc[2][r]);
                float ov = fast_sigmoid(acc[3][r]);
                float cv = fv * c[ci] + iv * gv;
                c[ci] = cv;
                hfull[ci] = f32_to_bf16(ov * fast_tanh(cv));
                if (t == TT - 1) cn[ci] = cv;
            }
        }
        grid_barrier(bar, bar + 1, SCAN_BLOCKS);

        // ---- stage h_full slab (16 rows x 1024) for phase 2 ----
        if (blockIdx.x < 16) {
            const int mtb = blockIdx.x >> 2;          // uniform: == phase-2 mt
            const unsigned short* gsrc = hfull + (size_t)mtb * 16 * HID;
            #pragma unroll
            for (int i = 0; i < 8; ++i) {
                int ci  = threadIdx.x + i * SCAN_THREADS;   // 0..2047 chunks
                int row = ci >> 7, col = (ci & 127) * 8;
                stage_chunk(gsrc + row * HID + col, smem + row * P2_STRIDE + col);
            }
            stage_wait();
        }
        __syncthreads();

        // ---- phase 2: h = h_full @ w_hr^T ; emit y[t] ----
        if (wave < 128) {
            const int pt = wave & 31;   // 0..31 proj tile
            const int mt = wave >> 5;   // 0..3 batch tile
            v8f acc = v8f{0};
            const unsigned short* arow = smem + m16 * P2_STRIDE;  // LDS A slab
            #pragma unroll 4
            for (int k0 = 0; k0 < HID; k0 += 32) {
                v16bf a = frag16(arow + k0 + abase, arow + k0 + 16 + abase);
                const unsigned short* brow =
                    whr + (size_t)(pt * 16 + m16) * HID + k0 + bbase;
                acc = wmma_bf16(a, frag16(brow, brow + 16), acc);
            }
            const int col = pt * 16 + m16;
            #pragma unroll
            for (int r = 0; r < 8; ++r) {
                const int n = mt * 16 + half * 8 + r;
                float v = acc[r];
                __builtin_nontemporal_store(v, &y[((size_t)t * NB + n) * PR + col]);
                hb[(size_t)n * PR + col] = f32_to_bf16(v);
                if (t == TT - 1) hn[(size_t)n * PR + col] = v;
            }
        }
        grid_barrier(bar, bar + 1, SCAN_BLOCKS);
    }
}

// ---------------- host launcher ----------------
extern "C" void kernel_launch(void* const* d_in, const int* in_sizes, int n_in,
                              void* d_out, int out_size, void* d_ws, size_t ws_size,
                              hipStream_t stream) {
    (void)in_sizes; (void)n_in; (void)out_size; (void)ws_size;
    const float* x   = (const float*)d_in[0];
    const float* h0  = (const float*)d_in[1];
    const float* c0  = (const float*)d_in[2];
    const float* wih = (const float*)d_in[3];
    const float* whh = (const float*)d_in[4];
    const float* b   = (const float*)d_in[5];
    const float* whr = (const float*)d_in[6];

    char* ws = (char*)d_ws;
    unsigned short* xb   = (unsigned short*)(ws + OFF_XB);
    unsigned short* wihb = (unsigned short*)(ws + OFF_WIH);
    unsigned short* whhb = (unsigned short*)(ws + OFF_WHH);
    unsigned short* whrb = (unsigned short*)(ws + OFF_WHR);
    unsigned short* xg   = (unsigned short*)(ws + OFF_XG);
    float*          cbuf = (float*)(ws + OFF_C);
    unsigned short* hb   = (unsigned short*)(ws + OFF_HB);
    unsigned short* hf   = (unsigned short*)(ws + OFF_HF);
    unsigned*       bar  = (unsigned*)(ws + OFF_BAR);

    float* y  = (float*)d_out;
    float* hn = y + (size_t)TT * NB * PR;
    float* cn = hn + (size_t)NB * PR;

    const int thr = 256;
    k_f32_to_bf16<<<(TT * NB * IN + thr - 1) / thr, thr, 0, stream>>>(x, xb, TT * NB * IN);
    k_f32_to_bf16<<<(4 * HID * IN + thr - 1) / thr, thr, 0, stream>>>(wih, wihb, 4 * HID * IN);
    k_f32_to_bf16<<<(4 * HID * PR + thr - 1) / thr, thr, 0, stream>>>(whh, whhb, 4 * HID * PR);
    k_f32_to_bf16<<<(PR * HID + thr - 1) / thr, thr, 0, stream>>>(whr, whrb, PR * HID);
    k_init_state<<<(NB * HID + thr - 1) / thr, thr, 0, stream>>>(h0, c0, hb, cbuf, bar);

    k_gates_gemm<<<2048, 256, 0, stream>>>(xb, wihb, b, xg);
    k_scan<<<SCAN_BLOCKS, SCAN_THREADS, 0, stream>>>(xg, whhb, whrb, hb, cbuf, hf,
                                                     y, hn, cn, bar);
}
